// MiTStage_635655160375
// MI455X (gfx1250) — matrix operand order repaired and checked by reading, not compiled, not run on hardware
//
#include <hip/hip_runtime.h>
#include <hip/hip_bf16.h>
#include <math.h>

typedef __bf16 bf16_t;
typedef __attribute__((ext_vector_type(16))) __bf16 v16bf;
typedef __attribute__((ext_vector_type(8)))  __bf16 v8bf;
typedef __attribute__((ext_vector_type(8)))  float  v8f;

// ---------------------------------------------------------------------------
// fp32 -> bf16 conversion (weights / activations)
// ---------------------------------------------------------------------------
__global__ void __launch_bounds__(256)
convert_f32_bf16_kernel(const float* __restrict__ in, bf16_t* __restrict__ out, int n) {
    int i = blockIdx.x * 256 + threadIdx.x;
    if (i < n) out[i] = (bf16_t)in[i];
}

// Convert [layers, rows, cols] fp32 -> [layers, padRows, cols] bf16, zero pad.
__global__ void __launch_bounds__(256)
convert_pad_kernel(const float* __restrict__ src, bf16_t* __restrict__ dst,
                   int rows, int cols, int padRows, int total) {
    int idx = blockIdx.x * 256 + threadIdx.x;
    if (idx >= total) return;
    int c  = idx % cols;
    int rr = (idx / cols) % padRows;
    int l  = idx / (cols * padRows);
    dst[idx] = (rr < rows) ? (bf16_t)src[((size_t)l * rows + rr) * cols + c]
                           : (bf16_t)0.0f;
}

// ---------------------------------------------------------------------------
// Fold BatchNorm into per-channel scale/shift:  y = v*s[e] + t[e]
// ---------------------------------------------------------------------------
__global__ void bn_affine_kernel(const float* __restrict__ g, const float* __restrict__ be,
                                 const float* __restrict__ mean, const float* __restrict__ var,
                                 float* __restrict__ scale, float* __restrict__ shift) {
    int e = threadIdx.x;   // 128 threads
    float s = g[e] * rsqrtf(var[e] + 1e-5f);
    scale[e] = s;
    shift[e] = be[e] - mean[e] * s;
}

// ---------------------------------------------------------------------------
// Front end: 2x2 avg pool -> reduce(16) -> relu -> span(9) -> 3x3 stride-2
// weighted gather over 64 input channels. Emits v (BN x 64) in bf16 for the
// proj GEMM. One thread per output pixel (b, ho, wo).
// ---------------------------------------------------------------------------
__global__ void __launch_bounds__(256)
frontend_kernel(const float* __restrict__ x,
                const float* __restrict__ rw, const float* __restrict__ rb,
                const float* __restrict__ sw, const float* __restrict__ sb,
                bf16_t* __restrict__ vbuf) {
    int idx = blockIdx.x * 256 + threadIdx.x;      // b*4096 + ho*64 + wo
    int wo  = idx & 63;
    int ho  = (idx >> 6) & 63;
    int b   = idx >> 12;
    const float* xb = x + (size_t)b * 64 * 128 * 128;

    float hid[16];
    #pragma unroll
    for (int o = 0; o < 16; ++o) hid[o] = rb[o];

    int h2 = ho * 2, w2 = wo * 2;
    for (int c = 0; c < 64; ++c) {
        const float* xc = xb + (size_t)c * 128 * 128;
        float pool = 0.25f * (xc[h2 * 128 + w2]       + xc[h2 * 128 + w2 + 1] +
                              xc[(h2 + 1) * 128 + w2] + xc[(h2 + 1) * 128 + w2 + 1]);
        #pragma unroll
        for (int o = 0; o < 16; ++o) hid[o] += rw[o * 64 + c] * pool;
    }
    #pragma unroll
    for (int o = 0; o < 16; ++o) hid[o] = fmaxf(hid[o], 0.f);

    float Kv[9];
    #pragma unroll
    for (int k = 0; k < 9; ++k) {
        float a = sb[k];
        #pragma unroll
        for (int o = 0; o < 16; ++o) a += sw[k * 16 + o] * hid[o];
        Kv[k] = a;
    }

    bf16_t* vout = vbuf + (size_t)idx * 64;
    for (int c = 0; c < 64; ++c) {
        const float* xc = xb + (size_t)c * 128 * 128;
        float v = 0.f;
        #pragma unroll
        for (int i = 0; i < 3; ++i)
            #pragma unroll
            for (int j = 0; j < 3; ++j) {
                int hh = h2 + i - 1, ww = w2 + j - 1;
                if (hh >= 0 && hh < 128 && ww >= 0 && ww < 128)
                    v += Kv[i * 3 + j] * xc[hh * 128 + ww];
            }
        vout[c] = (bf16_t)v;
    }
}

// ---------------------------------------------------------------------------
// bf16 WMMA GEMM:  C[M,N] = A[M,K] * W[N,K]^T  (fp32 accumulate)
// One wave = 32(M) x 16(N): two v_wmma per K-step sharing one B fragment.
// Block = 8 waves = 32(M) x 128(N). W must be padded to >= roundup16(N) rows
// so B-fragment loads are unconditional (no EXEC masking in the K-loop).
// Lane layouts (A/B k order is this kernel's choice; C/D per CDNA5 ISA):
//   A (16x32 bf16): lane row M=l&15; e -> K=k0+16h+e  (contiguous, matches B)
//   B (32x16 bf16): lane col N=l&15;  e -> K=k0+16h+e  (contiguous row of W)
//   C/D (16x16 f32): vgpr r -> M = r + 8*(lane>>4), N = lane&15
// Epilogue: optional per-N scale/bias, exact gelu, residual accumulate, bf16 copy.
// ---------------------------------------------------------------------------
__global__ void __launch_bounds__(256)
gemm_bf16_wmma_kernel(const bf16_t* __restrict__ A,
                      const bf16_t* __restrict__ W,
                      float* __restrict__ C,
                      bf16_t* __restrict__ Cbf,
                      int M, int N, int K,
                      const float* __restrict__ scale,
                      const float* __restrict__ bias,
                      int act, int accum) {
    const int wave = threadIdx.x >> 5;
    const int lane = threadIdx.x & 31;
    const int half = lane >> 4;
    const int l15  = lane & 31 & 15;
    const int mBase = blockIdx.x << 5;                  // 32 rows / block
    const int nBase = (blockIdx.y << 7) + (wave << 4);
    if (nBase >= N) return;                             // wave-uniform
    const int nOut = nBase + l15;

    const bf16_t* aRow0 = A + (size_t)(mBase + l15) * K + 16 * half;
    const bf16_t* aRow1 = aRow0 + (size_t)16 * K;
    const bf16_t* bRow  = W + (size_t)nOut * K + 16 * half;   // always in-bounds (padded)

    v8f acc0 = {}, acc1 = {};
    for (int k0 = 0; k0 < K; k0 += 32) {
        v8bf a00 = *(const v8bf*)(aRow0 + k0);
        v8bf a01 = *(const v8bf*)(aRow0 + k0 + 8);
        v8bf a10 = *(const v8bf*)(aRow1 + k0);
        v8bf a11 = *(const v8bf*)(aRow1 + k0 + 8);
        v16bf b  = *(const v16bf*)(bRow + k0);
        v16bf a0, a1;
        #pragma unroll
        for (int e = 0; e < 8; ++e) {
            a0[e] = a00[e]; a0[e + 8] = a01[e];
            a1[e] = a10[e]; a1[e + 8] = a11[e];
        }
        if (k0 + 32 < K) {                              // global_prefetch_b8 path
            __builtin_prefetch(aRow0 + k0 + 32, 0, 1);
            __builtin_prefetch(aRow1 + k0 + 32, 0, 1);
            __builtin_prefetch(bRow  + k0 + 32, 0, 1);
        }
        acc0 = __builtin_amdgcn_wmma_f32_16x16x32_bf16(false, a0, false, b,
                                                       (short)0, acc0, false, false);
        acc1 = __builtin_amdgcn_wmma_f32_16x16x32_bf16(false, a1, false, b,
                                                       (short)0, acc1, false, false);
    }
    if (nOut >= N) return;                              // store guard only

    const float sc = scale ? scale[nOut] : 1.f;
    const float bi = bias  ? bias[nOut]  : 0.f;
    #pragma unroll
    for (int r = 0; r < 8; ++r) {
        int   m0 = mBase + r + 8 * half;
        float v0 = acc0[r] * sc + bi;
        float v1 = acc1[r] * sc + bi;
        if (act == 1) {
            v0 = 0.5f * v0 * (1.f + erff(v0 * 0.70710678118f));
            v1 = 0.5f * v1 * (1.f + erff(v1 * 0.70710678118f));
        }
        size_t o0 = (size_t)m0 * N + nOut;
        size_t o1 = o0 + (size_t)16 * N;
        if (C) {
            if (accum) { v0 += C[o0]; v1 += C[o1]; }
            C[o0] = v0; C[o1] = v1;
        }
        if (Cbf) { Cbf[o0] = (bf16_t)v0; Cbf[o1] = (bf16_t)v1; }
    }
}

// ---------------------------------------------------------------------------
// RMSNorm over E=128 per token: one wave per token, 4 features per lane.
// ---------------------------------------------------------------------------
__global__ void __launch_bounds__(256)
rmsnorm_kernel(const float* __restrict__ t, const float* __restrict__ w,
               bf16_t* __restrict__ out, int rows) {
    int token = blockIdx.x * 8 + (threadIdx.x >> 5);
    int lane  = threadIdx.x & 31;
    if (token >= rows) return;
    const float* r = t + (size_t)token * 128;
    float v0 = r[lane], v1 = r[lane + 32], v2 = r[lane + 64], v3 = r[lane + 96];
    float s = v0 * v0 + v1 * v1 + v2 * v2 + v3 * v3;
    #pragma unroll
    for (int off = 16; off > 0; off >>= 1) s += __shfl_xor(s, off, 32);
    float inv = rsqrtf(s * (1.f / 128.f) + 1e-5f);
    bf16_t* o = out + (size_t)token * 128;
    o[lane]      = (bf16_t)(v0 * inv * w[lane]);
    o[lane + 32] = (bf16_t)(v1 * inv * w[lane + 32]);
    o[lane + 64] = (bf16_t)(v2 * inv * w[lane + 64]);
    o[lane + 96] = (bf16_t)(v3 * inv * w[lane + 96]);
}

// ---------------------------------------------------------------------------
// Causal width-4 conv along the sequence + bias + silu.
// xz is (BN,512); xc channels live at offset 0. Thread per (token, d).
// ---------------------------------------------------------------------------
__global__ void __launch_bounds__(256)
conv1d_silu_kernel(const float* __restrict__ xz,
                   const float* __restrict__ cw, const float* __restrict__ cb,
                   float* __restrict__ xcf, bf16_t* __restrict__ xcb, int Nseq) {
    int idx = blockIdx.x * 256 + threadIdx.x;   // BN*256
    int dch = idx & 255;
    int row = idx >> 8;
    int t   = row & (Nseq - 1);
    float acc = cb[dch];
    #pragma unroll
    for (int j = 0; j < 4; ++j) {
        int tt = t + j - 3;
        if (tt >= 0) acc += cw[dch * 4 + j] * xz[(size_t)(row + j - 3) * 512 + dch];
    }
    float s = acc / (1.f + __expf(-acc));       // silu
    xcf[(size_t)row * 256 + dch] = s;
    xcb[(size_t)row * 256 + dch] = (bf16_t)s;
}

// ---------------------------------------------------------------------------
// Fused selective scan: dt = softplus(dt_low . Wdt[d] + b[d]); per step
//   h = exp(dt*A)*h + (dt*u)*B ; y = sum_n h*C ; out = (y + u*D) * silu(z).
// Thread = one (b,d,n) state; 16-lane shuffle reduce over n. 512 waves total.
// dbl row layout: [0:8)=dt_low  [8:24)=B  [24:40)=C (uniform per block row).
// ---------------------------------------------------------------------------
__global__ void __launch_bounds__(256)
scan_kernel(const float* __restrict__ dbl, const float* __restrict__ xc,
            const float* __restrict__ xz, const float* __restrict__ A_log,
            const float* __restrict__ Dvec, const float* __restrict__ Wdt,
            const float* __restrict__ bdt, bf16_t* __restrict__ ybf, int Nseq) {
    int n   = threadIdx.x & 15;
    int g   = threadIdx.x >> 4;                 // 16 d-channels per block
    int b   = blockIdx.x >> 4;
    int dch = ((blockIdx.x & 15) << 4) + g;

    float Acoef = -__expf(A_log[dch * 16 + n]);
    float wv[8];
    #pragma unroll
    for (int r = 0; r < 8; ++r) wv[r] = Wdt[dch * 8 + r];
    float bias = bdt[dch];
    float Dd   = Dvec[dch];
    float h    = 0.f;

    size_t rowBase = (size_t)b * Nseq;
    for (int t = 0; t < Nseq; ++t) {
        size_t row = rowBase + t;
        const float* dl = dbl + row * 40;
        float dtl = bias;
        #pragma unroll
        for (int r = 0; r < 8; ++r) dtl += wv[r] * dl[r];
        float dt = (dtl > 20.f) ? dtl : __logf(1.f + __expf(dtl));   // softplus
        float u  = xc[row * 256 + dch];
        float dA = __expf(dt * Acoef);
        h = dA * h + (dt * u) * dl[8 + n];
        float y = h * dl[24 + n];
        y += __shfl_xor(y, 1, 16);
        y += __shfl_xor(y, 2, 16);
        y += __shfl_xor(y, 4, 16);
        y += __shfl_xor(y, 8, 16);
        if (n == 0) {
            float zv = xz[row * 512 + 256 + dch];
            float sz = zv / (1.f + __expf(-zv));
            ybf[row * 256 + dch] = (bf16_t)((y + u * Dd) * sz);
        }
    }
}

// ---------------------------------------------------------------------------
// (B,N,E) fp32 -> (B,E,H,W) output
// ---------------------------------------------------------------------------
__global__ void __launch_bounds__(256)
output_kernel(const float* __restrict__ t, float* __restrict__ out) {
    int idx = blockIdx.x * 256 + threadIdx.x;   // B*128*64*64
    int w = idx & 63;
    int h = (idx >> 6) & 63;
    int e = (idx >> 12) & 127;
    int b = idx >> 19;
    out[idx] = t[((size_t)b * 4096 + h * 64 + w) * 128 + e];
}

// ---------------------------------------------------------------------------
extern "C" void kernel_launch(void* const* d_in, const int* in_sizes, int n_in,
                              void* d_out, int out_size, void* d_ws, size_t ws_size,
                              hipStream_t stream) {
    (void)in_sizes; (void)n_in; (void)out_size; (void)ws_size;

    const float* x         = (const float*)d_in[0];
    const float* reduce_w  = (const float*)d_in[1];
    const float* reduce_b  = (const float*)d_in[2];
    const float* span_w    = (const float*)d_in[3];
    const float* span_b    = (const float*)d_in[4];
    const float* proj_w    = (const float*)d_in[5];
    const float* bn_gamma  = (const float*)d_in[6];
    const float* bn_beta   = (const float*)d_in[7];
    const float* bn_mean   = (const float*)d_in[8];
    const float* bn_var    = (const float*)d_in[9];
    const float* norm1_w   = (const float*)d_in[10];
    const float* in_proj_w = (const float*)d_in[11];
    const float* conv_w    = (const float*)d_in[12];
    const float* conv_b    = (const float*)d_in[13];
    const float* x_proj_w  = (const float*)d_in[14];
    const float* dt_proj_w = (const float*)d_in[15];
    const float* dt_proj_b = (const float*)d_in[16];
    const float* A_log     = (const float*)d_in[17];
    const float* Dvec      = (const float*)d_in[18];
    const float* out_proj_w= (const float*)d_in[19];
    const float* norm2_w   = (const float*)d_in[20];
    const float* fc1_w     = (const float*)d_in[21];
    const float* fc1_b     = (const float*)d_in[22];
    const float* fc2_w     = (const float*)d_in[23];
    const float* fc2_b     = (const float*)d_in[24];

    constexpr int Bb = 4, Nseq = 4096, BN = Bb * Nseq, E = 128, Dm = 256, HM = 512;

    char* p = (char*)d_ws;
    auto wsa = [&](size_t bytes) -> void* {
        void* r = (void*)p;
        p += (bytes + 255) & ~(size_t)255;
        return r;
    };
    float*  t_act = (float*) wsa((size_t)BN * E * 4);
    bf16_t* vbuf  = (bf16_t*)wsa((size_t)BN * 64 * 2);
    bf16_t* hn    = (bf16_t*)wsa((size_t)BN * E * 2);
    float*  xz    = (float*) wsa((size_t)BN * 512 * 4);
    float*  xcf   = (float*) wsa((size_t)BN * Dm * 4);
    bf16_t* xcb   = (bf16_t*)wsa((size_t)BN * Dm * 2);
    float*  dbl   = (float*) wsa((size_t)BN * 40 * 4);
    bf16_t* ybf   = (bf16_t*)wsa((size_t)BN * Dm * 2);
    bf16_t* h2    = (bf16_t*)wsa((size_t)BN * HM * 2);
    float*  bnsc  = (float*) wsa(512);
    float*  bnsh  = (float*) wsa(512);
    bf16_t* wproj = (bf16_t*)wsa((size_t)128 * 64 * 2);
    bf16_t* winp  = (bf16_t*)wsa((size_t)2 * 512 * 128 * 2);
    bf16_t* wxp   = (bf16_t*)wsa((size_t)2 * 48 * 256 * 2);   // padded 40 -> 48 rows
    bf16_t* wout  = (bf16_t*)wsa((size_t)2 * 128 * 256 * 2);
    bf16_t* wfc1  = (bf16_t*)wsa((size_t)2 * 512 * 128 * 2);
    bf16_t* wfc2  = (bf16_t*)wsa((size_t)2 * 128 * 512 * 2);

    auto cvt = [&](const float* src, bf16_t* dst, int n) {
        convert_f32_bf16_kernel<<<(n + 255) / 256, 256, 0, stream>>>(src, dst, n);
    };
    cvt(proj_w,     wproj, 128 * 64);
    cvt(in_proj_w,  winp,  2 * 512 * 128);
    cvt(out_proj_w, wout,  2 * 128 * 256);
    cvt(fc1_w,      wfc1,  2 * 512 * 128);
    cvt(fc2_w,      wfc2,  2 * 128 * 512);
    {   // x_proj: pad 40 rows -> 48 zero-filled rows so B loads need no guard
        int total = 2 * 48 * 256;
        convert_pad_kernel<<<(total + 255) / 256, 256, 0, stream>>>(
            x_proj_w, wxp, 40, 256, 48, total);
    }

    bn_affine_kernel<<<1, 128, 0, stream>>>(bn_gamma, bn_beta, bn_mean, bn_var, bnsc, bnsh);
    frontend_kernel<<<BN / 256, 256, 0, stream>>>(x, reduce_w, reduce_b, span_w, span_b, vbuf);

    auto gemm = [&](const bf16_t* Aa, const bf16_t* Ww, float* Cc, bf16_t* Cb,
                    int Nn, int Kk, const float* sc, const float* bi, int act, int accum) {
        dim3 grid(BN / 32, (Nn + 127) / 128);
        gemm_bf16_wmma_kernel<<<grid, 256, 0, stream>>>(Aa, Ww, Cc, Cb, BN, Nn, Kk, sc, bi, act, accum);
    };

    // proj (64->128) with folded BatchNorm epilogue -> token matrix t
    gemm(vbuf, wproj, t_act, nullptr, 128, 64, bnsc, bnsh, 0, 0);

    for (int i = 0; i < 2; ++i) {
        rmsnorm_kernel<<<BN / 8, 256, 0, stream>>>(t_act, norm1_w + i * E, hn, BN);
        gemm(hn, winp + (size_t)i * 512 * 128, xz, nullptr, 512, 128, nullptr, nullptr, 0, 0);
        conv1d_silu_kernel<<<(BN * Dm) / 256, 256, 0, stream>>>(
            xz, conv_w + i * Dm * 4, conv_b + i * Dm, xcf, xcb, Nseq);
        gemm(xcb, wxp + (size_t)i * 48 * 256, dbl, nullptr, 40, 256, nullptr, nullptr, 0, 0);
        scan_kernel<<<Bb * 16, 256, 0, stream>>>(
            dbl, xcf, xz, A_log + i * Dm * 16, Dvec + i * Dm,
            dt_proj_w + i * Dm * 8, dt_proj_b + i * Dm, ybf, Nseq);
        gemm(ybf, wout + (size_t)i * 128 * 256, t_act, nullptr, 128, 256, nullptr, nullptr, 0, 1);
        rmsnorm_kernel<<<BN / 8, 256, 0, stream>>>(t_act, norm2_w + i * E, hn, BN);
        gemm(hn, wfc1 + (size_t)i * 512 * 128, nullptr, h2, 512, 128, nullptr, fc1_b + i * HM, 1, 0);
        gemm(h2, wfc2 + (size_t)i * 128 * 512, t_act, nullptr, 128, 512, nullptr, fc2_b + i * E, 0, 1);
    }

    output_kernel<<<(Bb * E * 64 * 64) / 256, 256, 0, stream>>>(t_act, (float*)d_out);
}
